// LoRA_GRU_67817533604217
// MI455X (gfx1250) — matrix operand order repaired and checked by reading, not compiled, not run on hardware
//
#include <hip/hip_runtime.h>
#include <cstdint>
#include <cstddef>

// Problem constants (match reference).
#define T_DIM   128
#define B_DIM   256
#define IN_DIM  512
#define H_DIM   512
#define H3_DIM  1536
#define R_DIM   16
#define NA_DIM  8
#define ALPHA_V 1.0f

typedef __bf16 bf16_t;
typedef bf16_t v16bf __attribute__((ext_vector_type(16)));
typedef float  v8f   __attribute__((ext_vector_type(8)));
typedef int    v4i_t __attribute__((ext_vector_type(4)));

union FragBF { uint4 q[2]; v16bf v; };   // 16 bf16 = 32B = 8 VGPRs

// ---- CDNA5 async global->LDS path (guarded; host pass / older toolchains fall back) ----
#if defined(__AMDGCN__) && __has_builtin(__builtin_amdgcn_global_load_async_to_lds_b128)
#define HAVE_ASYNC_LDS 1
#else
#define HAVE_ASYNC_LDS 0
#endif

__device__ __forceinline__ void async_cp16(const unsigned short* g, unsigned short* l) {
#if HAVE_ASYNC_LDS
  // Signature (from compiler diagnostic): (v4i AS1*, v4i AS3*, imm offset, imm cpol).
  // AS1 pointer: flat global address is the AS1 address.
  // AS3 pointer: low 32 bits of the generic LDS address are the LDS offset (ISA aperture rule).
  __builtin_amdgcn_global_load_async_to_lds_b128(
      (__attribute__((address_space(1))) v4i_t*)(uintptr_t)g,
      (__attribute__((address_space(3))) v4i_t*)(uint32_t)(uintptr_t)l,
      0, 0);
#else
  *reinterpret_cast<uint4*>(l) = *reinterpret_cast<const uint4*>(g);
#endif
}

template <int N>
__device__ __forceinline__ void wait_async() {
#if HAVE_ASYNC_LDS
#if __has_builtin(__builtin_amdgcn_s_wait_asynccnt)
  __builtin_amdgcn_s_wait_asynccnt(N);
#else
  asm volatile("s_wait_asynccnt %0" ::"n"(N));
#endif
#endif
}

__device__ __forceinline__ unsigned short f2bf(float f) {
  // round-to-nearest-even fp32 -> bf16
  unsigned int u = __float_as_uint(f);
  unsigned int r = 0x7FFFu + ((u >> 16) & 1u);
  return (unsigned short)((u + r) >> 16);
}

__device__ __forceinline__ float dot16(const float4* __restrict__ a,
                                       const float4* __restrict__ b) {
  float s = 0.f;
#pragma unroll
  for (int i = 0; i < 4; ++i) {
    float4 x = a[i], y = b[i];
    s = fmaf(x.x, y.x, s); s = fmaf(x.y, y.y, s);
    s = fmaf(x.z, y.z, s); s = fmaf(x.w, y.w, s);
  }
  return s;
}

// ---------------- elementwise helpers ----------------

__global__ void cvt_f32_bf16_v4(const float4* __restrict__ in,
                                ushort4* __restrict__ out, int n4) {
  int i = blockIdx.x * blockDim.x + threadIdx.x;
  if (i < n4) {
    float4 v = in[i];
    ushort4 o;
    o.x = f2bf(v.x); o.y = f2bf(v.y); o.z = f2bf(v.z); o.w = f2bf(v.w);
    out[i] = o;
  }
}

__global__ void h_init(const float* __restrict__ h0, float* __restrict__ H,
                       unsigned short* __restrict__ Hbf, int n) {
  int i = blockIdx.x * blockDim.x + threadIdx.x;
  if (i < n) { float v = h0[i]; H[i] = v; Hbf[i] = f2bf(v); }
}

__global__ void copy_f32(const float* __restrict__ in, float* __restrict__ out, int n) {
  int i = blockIdx.x * blockDim.x + threadIdx.x;
  if (i < n) out[i] = in[i];
}

// XA[m,r] = sum_k X[m,k] * AL[agent[m%Bmod], r, k]   (fp32 VALU, tiny)
__global__ void lora_xa(const float* __restrict__ X,   // [M,K] f32
                        const float* __restrict__ AL,  // [NA,16,K] f32
                        const int* __restrict__ agent, // [B]
                        float* __restrict__ XA,        // [M,16]
                        int M, int K, int Bmod) {
  int idx = blockIdx.x * blockDim.x + threadIdx.x;
  if (idx >= M * 16) return;
  int m = idx >> 4, rr = idx & 15;
  int g = agent[m % Bmod];
  const float4* xp = reinterpret_cast<const float4*>(X + (size_t)m * K);
  const float4* ap = reinterpret_cast<const float4*>(AL + ((size_t)g * 16 + rr) * K);
  float s = 0.f;
  for (int k = 0; k < K / 4; ++k) {
    float4 xv = xp[k], av = ap[k];
    s = fmaf(xv.x, av.x, s); s = fmaf(xv.y, av.y, s);
    s = fmaf(xv.z, av.z, s); s = fmaf(xv.w, av.w, s);
  }
  XA[idx] = s;
}

// ---------------- WMMA GEMM: async-LDS staged, double buffered ----------------
// C[m,n] = sum_k Abf[m,k]*Bbf[n,k] + bias[n] + alpha * sum_r XA[m,r]*BL[agent[m%Bmod], n, r]
template <int WMT, int WNT, int WVM, int WVN>
__global__ __launch_bounds__(32 * WVM * WVN)
void gemm_bf16_lora(const unsigned short* __restrict__ Abf, // [M,K] bf16
                    const unsigned short* __restrict__ Bbf, // [N,K] bf16
                    const float* __restrict__ bias,         // [N]
                    const float* __restrict__ XA,           // [M,16]
                    const float* __restrict__ BL,           // [NA,N,16]
                    const int* __restrict__ agent,          // [Bmod]
                    float* __restrict__ C,                  // [M,N]
                    int M, int N, int K, int Bmod, float alpha) {
  constexpr int NTH = 32 * WVM * WVN;
  constexpr int BM  = WVM * WMT * 16;
  constexpr int BN  = WVN * WNT * 16;
  constexpr int KC  = 32;              // K-chunk per stage
  constexpr int AST = KC + 8;          // padded LDS row stride (elems) -> conflict-free frags
  constexpr int SPR = KC / 8;          // 16B segments per row (=4)
  constexpr int ASEG = (BM * SPR) / NTH;
  constexpr int BSEG = (BN * SPR) / NTH;
  static_assert((BM * SPR) % NTH == 0 && (BN * SPR) % NTH == 0, "seg split");

  __shared__ __align__(16) unsigned short As[2][BM * AST];
  __shared__ __align__(16) unsigned short Bs[2][BN * AST];

  const int lane = threadIdx.x & 31;
  const int wave = threadIdx.x >> 5;
  const int wm   = wave / WVN;
  const int wn   = wave - wm * WVN;
  const int r16  = lane & 15;
  const int kh   = lane >> 4;

  const int mblk = blockIdx.y * BM;
  const int nblk = blockIdx.x * BN;
  const int wmB  = wm * WMT * 16;      // wave tile origin inside block tile
  const int wnB  = wn * WNT * 16;

  const unsigned short* Ablk = Abf + (size_t)mblk * K;
  const unsigned short* Bblk = Bbf + (size_t)nblk * K;

  auto stage = [&](int buf, int kc) {
#pragma unroll
    for (int s = 0; s < ASEG; ++s) {
      int seg = threadIdx.x + s * NTH;
      int row = seg >> 2;            // SPR == 4
      int col = (seg & 3) * 8;       // element offset
      async_cp16(Ablk + (size_t)row * K + kc + col, &As[buf][row * AST + col]);
    }
#pragma unroll
    for (int s = 0; s < BSEG; ++s) {
      int seg = threadIdx.x + s * NTH;
      int row = seg >> 2;
      int col = (seg & 3) * 8;
      async_cp16(Bblk + (size_t)row * K + kc + col, &Bs[buf][row * AST + col]);
    }
  };

  v8f zero = {0.f, 0.f, 0.f, 0.f, 0.f, 0.f, 0.f, 0.f};
  v8f acc[WMT][WNT];
#pragma unroll
  for (int i = 0; i < WMT; ++i)
#pragma unroll
    for (int j = 0; j < WNT; ++j)
      acc[i][j] = zero;

  const int NIT = K / KC;
  stage(0, 0);

  for (int it = 0; it < NIT; ++it) {
    const int cur = it & 1;
    if (it + 1 < NIT) {
      stage(cur ^ 1, (it + 1) * KC);
      wait_async<ASEG + BSEG>();     // my stage-`it` copies have landed
    } else {
      wait_async<0>();
    }
    __syncthreads();                 // everyone's stage-`it` copies have landed

    FragBF a[WMT], b[WNT];
    // A frag (16x32): lane holds row (lane&15); K = {kh*8..+7, 16+kh*8..+7}
#pragma unroll
    for (int i = 0; i < WMT; ++i) {
      const unsigned short* p = &As[cur][(wmB + i * 16 + r16) * AST + kh * 8];
      a[i].q[0] = *reinterpret_cast<const uint4*>(p);
      a[i].q[1] = *reinterpret_cast<const uint4*>(p + 16);
    }
    // B frag (32x16): lane holds col (lane&15); K = {kh*8..+7, 16+kh*8..+7}
#pragma unroll
    for (int j = 0; j < WNT; ++j) {
      const unsigned short* p = &Bs[cur][(wnB + j * 16 + r16) * AST + kh * 8];
      b[j].q[0] = *reinterpret_cast<const uint4*>(p);
      b[j].q[1] = *reinterpret_cast<const uint4*>(p + 16);
    }
#pragma unroll
    for (int i = 0; i < WMT; ++i)
#pragma unroll
      for (int j = 0; j < WNT; ++j)
        acc[i][j] = __builtin_amdgcn_wmma_f32_16x16x32_bf16(
            false, a[i].v, false, b[j].v, (short)0, acc[i][j], false, false);

    __syncthreads();                 // buffer `cur` may be re-targeted next iter
  }

  // Epilogue: C/D layout — lane holds col n=r16, rows m = v + 8*kh
#pragma unroll
  for (int i = 0; i < WMT; ++i) {
#pragma unroll
    for (int v = 0; v < 8; ++v) {
      const int m = mblk + wmB + i * 16 + kh * 8 + v;
      const int g = agent[m % Bmod];
      const float4* xa = reinterpret_cast<const float4*>(XA + (size_t)m * 16);
      float4 xr[4] = {xa[0], xa[1], xa[2], xa[3]};
#pragma unroll
      for (int j = 0; j < WNT; ++j) {
        const int n = nblk + wnB + j * 16 + r16;
        const float4* bl =
            reinterpret_cast<const float4*>(BL + ((size_t)g * N + n) * 16);
        float d = dot16(xr, bl);
        C[(size_t)m * N + n] = acc[i][j][v] + bias[n] + alpha * d;
      }
    }
  }
}

// ---------------- GRU gate fusion ----------------
__global__ void gru_gate(const float* __restrict__ XOt,    // [B,3H] slice at step t
                         const float* __restrict__ HO,     // [B,3H]
                         float* __restrict__ H,            // [B,H] fp32 state (in/out)
                         unsigned short* __restrict__ Hbf, // [B,H] bf16 state (out)
                         float* __restrict__ outT,         // out[t] = [B,H]
                         int BH, int Hdim) {
  int idx = blockIdx.x * blockDim.x + threadIdx.x;
  if (idx >= BH) return;
  int b = idx / Hdim;
  int j = idx - b * Hdim;
  size_t base = (size_t)b * (3 * Hdim) + j;
  float rx = XOt[base], zx = XOt[base + Hdim], nx = XOt[base + 2 * Hdim];
  float rh = HO[base],  zh = HO[base + Hdim],  nh = HO[base + 2 * Hdim];
  float r = 1.f / (1.f + expf(-(rx + rh)));
  float z = 1.f / (1.f + expf(-(zx + zh)));
  float n = tanhf(nx + r * nh);
  float h = H[idx];
  float hn = (1.f - z) * n + z * h;
  outT[idx] = hn;
  H[idx] = hn;
  Hbf[idx] = f2bf(hn);
}

// ---------------- launcher ----------------
extern "C" void kernel_launch(void* const* d_in, const int* in_sizes, int n_in,
                              void* d_out, int out_size, void* d_ws, size_t ws_size,
                              hipStream_t stream) {
  const float* x   = (const float*)d_in[0];   // [T,B,IN]
  const float* h0  = (const float*)d_in[1];   // [1,B,H]
  const float* Wx  = (const float*)d_in[2];   // [3H,IN]
  const float* bx  = (const float*)d_in[3];   // [3H]
  const float* Ax  = (const float*)d_in[4];   // [NA,R,IN]
  const float* Bx  = (const float*)d_in[5];   // [NA,3H,R]
  const float* Wh  = (const float*)d_in[6];   // [3H,H]
  const float* bh  = (const float*)d_in[7];   // [3H]
  const float* Ah  = (const float*)d_in[8];   // [NA,R,H]
  const float* Bh  = (const float*)d_in[9];   // [NA,3H,R]
  const int* agent = (const int*)d_in[10];    // [B]
  float* out = (float*)d_out;                 // [T,B,H] ++ [1,B,H]

  const int M = T_DIM * B_DIM;                // 32768

  // workspace carve-out (256B aligned)
  char* ws = (char*)d_ws;
  auto carve = [&](size_t bytes) -> char* {
    char* p = ws;
    ws += (bytes + 255) & ~(size_t)255;
    return p;
  };
  unsigned short* Xbf  = (unsigned short*)carve((size_t)M * IN_DIM * 2);
  unsigned short* Wxbf = (unsigned short*)carve((size_t)H3_DIM * IN_DIM * 2);
  unsigned short* Whbf = (unsigned short*)carve((size_t)H3_DIM * H_DIM * 2);
  float* XAx  = (float*)carve((size_t)M * R_DIM * 4);
  float* XO   = (float*)carve((size_t)M * H3_DIM * 4);
  float* Hcur = (float*)carve((size_t)B_DIM * H_DIM * 4);
  unsigned short* Hbf = (unsigned short*)carve((size_t)B_DIM * H_DIM * 2);
  float* HA = (float*)carve((size_t)B_DIM * R_DIM * 4);
  float* HO = (float*)carve((size_t)B_DIM * H3_DIM * 4);

  const int TPB = 256;
  auto blks = [](int n, int t) { return (n + t - 1) / t; };

  // 1) fp32 -> bf16 staging (vectorized x4)
  cvt_f32_bf16_v4<<<blks(M * IN_DIM / 4, TPB), TPB, 0, stream>>>(
      (const float4*)x, (ushort4*)Xbf, M * IN_DIM / 4);
  cvt_f32_bf16_v4<<<blks(H3_DIM * IN_DIM / 4, TPB), TPB, 0, stream>>>(
      (const float4*)Wx, (ushort4*)Wxbf, H3_DIM * IN_DIM / 4);
  cvt_f32_bf16_v4<<<blks(H3_DIM * H_DIM / 4, TPB), TPB, 0, stream>>>(
      (const float4*)Wh, (ushort4*)Whbf, H3_DIM * H_DIM / 4);

  // 2) x-path LoRA projection XA = x . Ax[g]^T
  lora_xa<<<blks(M * R_DIM, TPB), TPB, 0, stream>>>(x, Ax, agent, XAx, M, IN_DIM, B_DIM);

  // 3) hoisted x-path GEMM: XO = Xbf@Wx^T + bx + alpha*XA.Bx[g]^T
  //    128x128 block tile (8 waves of 64x32), async-LDS double buffered
  gemm_bf16_lora<4, 2, 2, 4>
      <<<dim3(H3_DIM / 128, M / 128), 256, 0, stream>>>(
          Xbf, Wxbf, bx, XAx, Bx, agent, XO, M, H3_DIM, IN_DIM, B_DIM, ALPHA_V);

  // 4) init recurrent state
  h_init<<<blks(B_DIM * H_DIM, TPB), TPB, 0, stream>>>(h0, Hcur, Hbf, B_DIM * H_DIM);

  // 5) recurrence: 128 sequential steps of (LoRA proj, WMMA GEMM, gates)
  for (int t = 0; t < T_DIM; ++t) {
    lora_xa<<<blks(B_DIM * R_DIM, TPB), TPB, 0, stream>>>(
        Hcur, Ah, agent, HA, B_DIM, H_DIM, B_DIM);
    //    64x128 block tile (8 waves of 32x32) -> 48 blocks at M=256
    gemm_bf16_lora<2, 2, 2, 4>
        <<<dim3(H3_DIM / 128, B_DIM / 64), 256, 0, stream>>>(
            Hbf, Whbf, bh, HA, Bh, agent, HO, B_DIM, H3_DIM, H_DIM, B_DIM, ALPHA_V);
    gru_gate<<<blks(B_DIM * H_DIM, TPB), TPB, 0, stream>>>(
        XO + (size_t)t * B_DIM * H3_DIM, HO, Hcur, Hbf,
        out + (size_t)t * B_DIM * H_DIM, B_DIM * H_DIM, H_DIM);
  }

  // 6) h_last tail
  copy_f32<<<blks(B_DIM * H_DIM, TPB), TPB, 0, stream>>>(
      Hcur, out + (size_t)T_DIM * B_DIM * H_DIM, B_DIM * H_DIM);
}
